// DCTAttention_71038759076382
// MI455X (gfx1250) — matrix operand order repaired and checked by reading, not compiled, not run on hardware
//
#include <hip/hip_runtime.h>
#include <hip/hip_bf16.h>

#define B_  8
#define H_  12
#define N_  2048
#define D_  64
#define M_  256
#define BH_ 96

typedef __attribute__((ext_vector_type(16))) __bf16 v16bf;
typedef __attribute__((ext_vector_type(8)))  __bf16 bf16x8;
typedef __attribute__((ext_vector_type(4)))  __bf16 bf16x4;
typedef __attribute__((ext_vector_type(8)))  float  v8f;

union Frag {
    v16bf  v;
    bf16x8 h[2];
    bf16x4 q[4];
};

// ---- WMMA fragment loaders -------------------------------------------------
// A-matrix (16x32 bf16): lane m (&15) holds row m; K = k0 + hi*16 + {0..15}   (hi = lane>=16)
__device__ __forceinline__ v16bf load_a_g(const __bf16* base, int stride, int row,
                                          int k0, int lane) {
    const int hi = (lane >> 4) & 1;
    const __bf16* p = base + (size_t)row * stride + k0 + hi * 16;
    Frag f;
    f.h[0] = *(const bf16x8*)(p);
    f.h[1] = *(const bf16x8*)(p + 8);
    return f.v;
}

// B-matrix (32x16 bf16): lane holds column n = lane&15; K = k0 + hi*16 + {0..15}
// (source stored row-per-column: base[row=n][k])
__device__ __forceinline__ v16bf load_b_g(const __bf16* base, int stride, int row,
                                          int k0, int lane) {
    const int hi = (lane >> 4) & 1;
    const __bf16* p = base + (size_t)row * stride + k0 + hi * 16;
    Frag f;
    f.h[0] = *(const bf16x8*)(p);
    f.h[1] = *(const bf16x8*)(p + 8);
    return f.v;
}

// Same as load_b_g but from LDS with only-8B-aligned rows -> b64 pairs
__device__ __forceinline__ v16bf load_b_lds(const __bf16* base, int stride, int row,
                                            int k0, int lane) {
    const int hi = (lane >> 4) & 1;
    const __bf16* p = base + row * stride + k0 + hi * 16;
    Frag f;
    f.q[0] = *(const bf16x4*)(p);
    f.q[1] = *(const bf16x4*)(p + 4);
    f.q[2] = *(const bf16x4*)(p + 8);
    f.q[3] = *(const bf16x4*)(p + 12);
    return f.v;
}

__device__ __forceinline__ v8f wmma_bf16(v16bf a, v16bf b, v8f c) {
    return __builtin_amdgcn_wmma_f32_16x16x32_bf16(false, a, false, b, (short)0, c,
                                                   false, false);
}

// ---- Kernel 0: convert DCT matrix to bf16 (row-major + transposed) ---------
__global__ __launch_bounds__(256) void k_prep_dct(const float* __restrict__ Qdct,
                                                  __bf16* __restrict__ QdctB,
                                                  __bf16* __restrict__ QdctT) {
    const int idx0 = (blockIdx.x * 256 + threadIdx.x) * 8;
#pragma unroll
    for (int j = 0; j < 8; ++j) {
        const int idx = idx0 + j;
        const int m = idx >> 11;          // row in [0,256)
        const int n = idx & (N_ - 1);     // col in [0,2048)
        const __bf16 v = (__bf16)Qdct[idx];
        QdctB[idx] = v;
        QdctT[(size_t)n * M_ + m] = v;
    }
}

// ---- Kernel 1: DCT projection  Yd[256x64] = QdctB[256x2048] * scaled(X)[2048x64]
// One WG per (head, tensor). 8 waves; wave owns 32 output rows (2x4 WMMA tiles).
// K-tile = 64 (two 32-wide sub-tiles), register-prefetch double buffering.
__global__ __launch_bounds__(256) void k_dct_proj(const float* __restrict__ Q,
                                                  const float* __restrict__ K,
                                                  const float* __restrict__ V,
                                                  const float* __restrict__ mask,
                                                  const __bf16* __restrict__ QdctB,
                                                  __bf16* __restrict__ Qd,
                                                  __bf16* __restrict__ Kd,
                                                  __bf16* __restrict__ Vd) {
    __shared__ __align__(16) __bf16 Bt[2][64 * 36];  // transposed B tiles [d][k]

    const int hd = blockIdx.x;   // head 0..95
    const int t  = blockIdx.y;   // 0=Q 1=K 2=V
    const int b  = hd / H_;
    const int tid  = threadIdx.x;
    const int lane = tid & 31;
    const int wave = tid >> 5;
    const float scale = 0.35355339059327373f;  // 64^-0.25

    const float* X = (t == 0 ? Q : (t == 1 ? K : V)) + (size_t)hd * N_ * D_;
    __bf16*      Y = (t == 0 ? Qd : (t == 1 ? Kd : Vd)) + (size_t)hd * M_ * D_;
    const float* mrow = mask + (size_t)b * N_;
    // branchless per-row scale: s = sA + sB * mask[row]
    const float sA = (t == 0) ? scale : 0.f;
    const float sB = (t == 0) ? 0.f : (t == 1 ? scale : 1.f);

    v8f acc[2][4];
#pragma unroll
    for (int i = 0; i < 2; ++i)
#pragma unroll
        for (int c = 0; c < 4; ++c) acc[i][c] = (v8f){};

    float4 pf[4];
    float  sc[4];
    auto prefetch = [&](int kt) {
#pragma unroll
        for (int i = 0; i < 4; ++i) {
            const int idx = tid + i * 256;
            const int kk  = idx >> 4;         // 0..63
            const int c4  = (idx & 15) << 2;  // 0..60
            pf[i] = *(const float4*)(X + (size_t)(kt + kk) * D_ + c4);
            sc[i] = sA + sB * mrow[kt + kk];
        }
    };
    prefetch(0);

    const int mrow0 = wave * 32;
    for (int kt = 0; kt < N_; kt += 64) {
        // store prefetched tile to LDS (converted, transposed)
#pragma unroll
        for (int i = 0; i < 4; ++i) {
            const int idx = tid + i * 256;
            const int kk  = idx >> 4;
            const int c4  = (idx & 15) << 2;
            __bf16* bp = &Bt[kk >> 5][0];
            const int k32 = kk & 31;
            const float sv = sc[i];
            bp[(c4 + 0) * 36 + k32] = (__bf16)(pf[i].x * sv);
            bp[(c4 + 1) * 36 + k32] = (__bf16)(pf[i].y * sv);
            bp[(c4 + 2) * 36 + k32] = (__bf16)(pf[i].z * sv);
            bp[(c4 + 3) * 36 + k32] = (__bf16)(pf[i].w * sv);
        }
        __syncthreads();

        // issue next tile's global loads; they overlap the WMMAs below
        prefetch((kt + 64) & (N_ - 1));  // wraps at the end (data unused)

        const v16bf a00 = load_a_g(QdctB, N_, mrow0 + (lane & 15), kt, lane);
        const v16bf a10 = load_a_g(QdctB, N_, mrow0 + 16 + (lane & 15), kt, lane);
        const v16bf a01 = load_a_g(QdctB, N_, mrow0 + (lane & 15), kt + 32, lane);
        const v16bf a11 = load_a_g(QdctB, N_, mrow0 + 16 + (lane & 15), kt + 32, lane);
#pragma unroll
        for (int c = 0; c < 4; ++c) {
            const v16bf b0 = load_b_lds(&Bt[0][0], 36, c * 16 + (lane & 15), 0, lane);
            acc[0][c] = wmma_bf16(a00, b0, acc[0][c]);
            acc[1][c] = wmma_bf16(a10, b0, acc[1][c]);
            const v16bf b1 = load_b_lds(&Bt[1][0], 36, c * 16 + (lane & 15), 0, lane);
            acc[0][c] = wmma_bf16(a01, b1, acc[0][c]);
            acc[1][c] = wmma_bf16(a11, b1, acc[1][c]);
        }
        __syncthreads();
    }

    const int hi = (lane >> 4) & 1;
#pragma unroll
    for (int i = 0; i < 2; ++i)
#pragma unroll
        for (int c = 0; c < 4; ++c)
#pragma unroll
            for (int v = 0; v < 8; ++v) {
                const int m = mrow0 + i * 16 + hi * 8 + v;
                const int n = c * 16 + (lane & 15);
                Y[(size_t)m * D_ + n] = (__bf16)acc[i][c][v];
            }
}

// ---- Kernel 2: energy = Qd*Kd^T (256x256, K=64) + row softmax -> P (bf16)
// One WG per head; 16 waves, each owns 16 rows (16 accumulator tiles).
__global__ __launch_bounds__(512) void k_attn_softmax(const __bf16* __restrict__ Qd,
                                                      const __bf16* __restrict__ Kd,
                                                      __bf16* __restrict__ P) {
    const int hd   = blockIdx.x;
    const int tid  = threadIdx.x;
    const int lane = tid & 31;
    const int wave = tid >> 5;  // 0..15
    const __bf16* Qh = Qd + (size_t)hd * M_ * D_;
    const __bf16* Kh = Kd + (size_t)hd * M_ * D_;
    __bf16*       Ph = P + (size_t)hd * M_ * M_;
    const int r0 = wave * 16;

    const v16bf a0 = load_a_g(Qh, D_, r0 + (lane & 15), 0, lane);
    const v16bf a1 = load_a_g(Qh, D_, r0 + (lane & 15), 32, lane);

    v8f acc[16];
#pragma unroll
    for (int t = 0; t < 16; ++t) {
        const v16bf b0 = load_b_g(Kh, D_, t * 16 + (lane & 15), 0, lane);
        const v16bf b1 = load_b_g(Kh, D_, t * 16 + (lane & 15), 32, lane);
        v8f c = (v8f){};
        c = wmma_bf16(a0, b0, c);
        c = wmma_bf16(a1, b1, c);
        acc[t] = c;
    }

    // row-wise softmax: row = r0 + hi*8 + v ; cols spread over (tile t, lane&15)
    float inv[8];
#pragma unroll
    for (int v = 0; v < 8; ++v) {
        float m = -1e30f;
#pragma unroll
        for (int t = 0; t < 16; ++t) m = fmaxf(m, acc[t][v]);
        m = fmaxf(m, __shfl_xor(m, 1));
        m = fmaxf(m, __shfl_xor(m, 2));
        m = fmaxf(m, __shfl_xor(m, 4));
        m = fmaxf(m, __shfl_xor(m, 8));
        float s = 0.f;
#pragma unroll
        for (int t = 0; t < 16; ++t) {
            const float e = __expf(acc[t][v] - m);
            acc[t][v] = e;
            s += e;
        }
        s += __shfl_xor(s, 1);
        s += __shfl_xor(s, 2);
        s += __shfl_xor(s, 4);
        s += __shfl_xor(s, 8);
        inv[v] = 1.f / s;
    }

    const int hi = (lane >> 4) & 1;
#pragma unroll
    for (int t = 0; t < 16; ++t)
#pragma unroll
        for (int v = 0; v < 8; ++v) {
            const int m = r0 + hi * 8 + v;
            const int n = t * 16 + (lane & 15);
            Ph[(size_t)m * M_ + n] = (__bf16)(acc[t][v] * inv[v]);
        }
}

// ---- Kernel 3: ctx[256x64] = P[256x256] * Vd[256x64] -----------------------
__global__ __launch_bounds__(256) void k_ctx(const __bf16* __restrict__ P,
                                             const __bf16* __restrict__ Vd,
                                             __bf16* __restrict__ Ctx) {
    __shared__ __align__(16) __bf16 VdT[64 * 260];  // [d][m], padded rows

    const int hd   = blockIdx.x;
    const int tid  = threadIdx.x;
    const int lane = tid & 31;
    const int wave = tid >> 5;
    const __bf16* Ph = P + (size_t)hd * M_ * M_;
    const __bf16* Vh = Vd + (size_t)hd * M_ * D_;
    __bf16*       Ch = Ctx + (size_t)hd * M_ * D_;

#pragma unroll
    for (int r0 = 0; r0 < M_; r0 += 32) {
        const int row = r0 + (tid >> 3);
        const int c0  = (tid & 7) * 8;
        const bf16x8 v = *(const bf16x8*)(Vh + (size_t)row * D_ + c0);
#pragma unroll
        for (int j = 0; j < 8; ++j) VdT[(c0 + j) * 260 + row] = v[j];
    }
    __syncthreads();

    v8f acc[2][4];
#pragma unroll
    for (int i = 0; i < 2; ++i)
#pragma unroll
        for (int c = 0; c < 4; ++c) acc[i][c] = (v8f){};

    const int mrow0 = wave * 32;
#pragma unroll
    for (int ks = 0; ks < 8; ++ks) {
        const int k0 = ks * 32;
        const v16bf a0 = load_a_g(Ph, M_, mrow0 + (lane & 15), k0, lane);
        const v16bf a1 = load_a_g(Ph, M_, mrow0 + 16 + (lane & 15), k0, lane);
#pragma unroll
        for (int c = 0; c < 4; ++c) {
            const v16bf bf = load_b_lds(VdT, 260, c * 16 + (lane & 15), k0, lane);
            acc[0][c] = wmma_bf16(a0, bf, acc[0][c]);
            acc[1][c] = wmma_bf16(a1, bf, acc[1][c]);
        }
    }

    const int hi = (lane >> 4) & 1;
#pragma unroll
    for (int i = 0; i < 2; ++i)
#pragma unroll
        for (int c = 0; c < 4; ++c)
#pragma unroll
            for (int v = 0; v < 8; ++v) {
                const int m = mrow0 + i * 16 + hi * 8 + v;
                const int n = c * 16 + (lane & 15);
                Ch[(size_t)m * D_ + n] = (__bf16)acc[i][c][v];
            }
}

// ---- Kernel 4: x[2048x64] = QdctT[2048x256] * ctx[256x64]  (fp32 output) ---
__global__ __launch_bounds__(256) void k_final(const __bf16* __restrict__ QdctT,
                                               const __bf16* __restrict__ Ctx,
                                               float* __restrict__ Out) {
    __shared__ __align__(16) __bf16 CtxT[64 * 260];

    const int hd    = blockIdx.x;
    const int strip = blockIdx.y;  // 0..7 (256-row strips of N)
    const int tid   = threadIdx.x;
    const int lane  = tid & 31;
    const int wave  = tid >> 5;
    const __bf16* Ch = Ctx + (size_t)hd * M_ * D_;
    float*        Oh = Out + (size_t)hd * N_ * D_;

#pragma unroll
    for (int r0 = 0; r0 < M_; r0 += 32) {
        const int row = r0 + (tid >> 3);
        const int c0  = (tid & 7) * 8;
        const bf16x8 v = *(const bf16x8*)(Ch + (size_t)row * D_ + c0);
#pragma unroll
        for (int j = 0; j < 8; ++j) CtxT[(c0 + j) * 260 + row] = v[j];
    }
    __syncthreads();

    v8f acc[2][4];
#pragma unroll
    for (int i = 0; i < 2; ++i)
#pragma unroll
        for (int c = 0; c < 4; ++c) acc[i][c] = (v8f){};

    const int n0 = strip * 256 + wave * 32;
#pragma unroll
    for (int ks = 0; ks < 8; ++ks) {
        const int k0 = ks * 32;
        const v16bf a0 = load_a_g(QdctT, M_, n0 + (lane & 15), k0, lane);
        const v16bf a1 = load_a_g(QdctT, M_, n0 + 16 + (lane & 15), k0, lane);
#pragma unroll
        for (int c = 0; c < 4; ++c) {
            const v16bf bf = load_b_lds(CtxT, 260, c * 16 + (lane & 15), k0, lane);
            acc[0][c] = wmma_bf16(a0, bf, acc[0][c]);
            acc[1][c] = wmma_bf16(a1, bf, acc[1][c]);
        }
    }

    const int hi = (lane >> 4) & 1;
#pragma unroll
    for (int i = 0; i < 2; ++i)
#pragma unroll
        for (int c = 0; c < 4; ++c)
#pragma unroll
            for (int v = 0; v < 8; ++v) {
                const int n = n0 + i * 16 + hi * 8 + v;
                const int d = c * 16 + (lane & 15);
                Oh[(size_t)n * D_ + d] = acc[i][c][v];
            }
}

// ---- Host launcher ---------------------------------------------------------
extern "C" void kernel_launch(void* const* d_in, const int* in_sizes, int n_in,
                              void* d_out, int out_size, void* d_ws, size_t ws_size,
                              hipStream_t stream) {
    const float* Q    = (const float*)d_in[0];
    const float* K    = (const float*)d_in[1];
    const float* V    = (const float*)d_in[2];
    const float* mask = (const float*)d_in[3];
    const float* Qdct = (const float*)d_in[4];
    float* Out = (float*)d_out;

    char* ws = (char*)d_ws;
    size_t off = 0;
    __bf16* QdctB = (__bf16*)(ws + off); off += (size_t)M_ * N_ * 2;   // 1 MB
    __bf16* QdctT = (__bf16*)(ws + off); off += (size_t)N_ * M_ * 2;   // 1 MB
    __bf16* Qd    = (__bf16*)(ws + off); off += (size_t)BH_ * M_ * D_ * 2;
    __bf16* Kd    = (__bf16*)(ws + off); off += (size_t)BH_ * M_ * D_ * 2;
    __bf16* Vd    = (__bf16*)(ws + off); off += (size_t)BH_ * M_ * D_ * 2;
    __bf16* P     = (__bf16*)(ws + off); off += (size_t)BH_ * M_ * M_ * 2;
    __bf16* Ctx   = (__bf16*)(ws + off); off += (size_t)BH_ * M_ * D_ * 2;

    k_prep_dct<<<256, 256, 0, stream>>>(Qdct, QdctB, QdctT);
    k_dct_proj<<<dim3(BH_, 3), 256, 0, stream>>>(Q, K, V, mask, QdctB, Qd, Kd, Vd);
    k_attn_softmax<<<BH_, 512, 0, stream>>>(Qd, Kd, P);
    k_ctx<<<BH_, 256, 0, stream>>>(P, Vd, Ctx);
    k_final<<<dim3(BH_, 8), 256, 0, stream>>>(QdctT, Ctx, Out);
}
